// LearnedPerformerAttention_31086973288489
// MI455X (gfx1250) — compile-verified, matrix operand order checked by reading, not run on hardware
//
#include <hip/hip_runtime.h>
#include <hip/hip_bf16.h>
#include <type_traits>

typedef __attribute__((ext_vector_type(16))) _Float16 v16h;
typedef __attribute__((ext_vector_type(4)))  _Float16 h4;
typedef __attribute__((ext_vector_type(8)))  float    v8f;
typedef __attribute__((ext_vector_type(4)))  float    f4;

// Problem constants (fixed by the reference)
constexpr int Bb = 4;
constexpr int Lh = 4096;
constexpr int Eh = 1024;
constexpr int Hh = 16;
constexpr int Dh = 64;     // head dim
constexpr int Mk = 256;    // kernel dim

// GEMM tiling: 256 threads = 8 waves (wave32), waves 4(m) x 2(n).
// Block tile 128x128, K-step 64. Each wave: 32x64 = 2x4 WMMA 16x16 accums,
// 16 v_wmma per K-tile per wave between barriers.
constexpr int BM   = 128;
constexpr int BN   = 128;
constexpr int BK   = 64;
constexpr int AROW = BK + 8;   // padded LDS row (halves) -> bank spread, 16B aligned
constexpr int BROW = BK + 8;
constexpr int ASLOT = BM * BK / (4 * 256);   // 8 half4/float4 per thread
constexpr int BSLOT = BK * BN / (4 * 256);   // 8 strips per thread

static_assert(BM == 128 && BN == 128 && BK == 64, "index math assumes these");

__device__ __forceinline__ h4 to_h4(f4 v) {
    h4 r; r.x = (_Float16)v.x; r.y = (_Float16)v.y; r.z = (_Float16)v.z; r.w = (_Float16)v.w;
    return r;
}
__device__ __forceinline__ h4 to_h4(h4 v) { return v; }

// ACT: 0 = none, 1 = elu(x)+1  (x>0 ? x+1 : exp(x))
template <typename TA, typename TC, bool TRANSA, int ACT>
__global__ __launch_bounds__(256)
void gemm_wmma_k(const TA* __restrict__ A, long aOuter, long aInner, int lda,
                 const float* __restrict__ Bm, long bOuter, long bInner, int ldb,
                 const float* __restrict__ bias,
                 TC* __restrict__ C, long cOuter, long cInner, int ldc,
                 int innerN, int Msz, int Nsz, int Ksz)
{
    const int z  = blockIdx.z;
    const int zo = z / innerN;
    const int zi = z % innerN;
    A  += (long)zo * aOuter + (long)zi * aInner;
    Bm += (long)zo * bOuter + (long)zi * bInner;
    C  += (long)zo * cOuter + (long)zi * cInner;

    const int tid  = threadIdx.x;
    const int lane = tid & 31;
    const int wave = tid >> 5;
    const int wm   = wave >> 1;            // 0..3  (m)
    const int wn   = wave & 1;             // 0..1  (n)
    const int hi   = lane >> 4;            // lane half: 0 or 1
    const int lmn  = lane & 15;            // 0..15

    const int blockM = blockIdx.y * BM;
    const int blockN = blockIdx.x * BN;

    __shared__ _Float16 sA[2][BM * AROW];  // [m][k], f16
    __shared__ _Float16 sB[2][BN * BROW];  // transposed [n][k], f16

    using AVec = typename std::conditional<sizeof(TA) == 4, f4, h4>::type;
    AVec aR[ASLOT];
    f4   bR[BSLOT];

    v8f acc[2][4] = {};

    // ---- per-thread staging index helpers (compile-time shapes) ----
    auto loadA = [&](int k0) {
        #pragma unroll
        for (int j = 0; j < ASLOT; ++j) {
            const int s = tid + j * 256;
            if (TRANSA) {                       // stored [K][M], contiguous in m
                const int kk = s >> 5;          // s / (BM/4)
                const int m  = (s & 31) * 4;
                aR[j] = *(const AVec*)(A + (long)(k0 + kk) * lda + (blockM + m));
            } else {                            // stored [M][K], contiguous in k
                const int m = s >> 4;           // s / (BK/4)
                const int k = (s & 15) * 4;
                aR[j] = *(const AVec*)(A + (long)(blockM + m) * lda + (k0 + k));
            }
        }
    };
    auto loadB = [&](int k0) {                  // 4 coalesced b32 down K per strip
        #pragma unroll
        for (int j = 0; j < BSLOT; ++j) {
            const int s  = tid + j * 256;
            const int n  = s & (BN - 1);
            const int k  = (s >> 7) * 4;
            const int gn = blockN + n;
            f4 v = {0.0f, 0.0f, 0.0f, 0.0f};
            if (gn < Nsz) {
                const float* bp = Bm + (long)(k0 + k) * ldb + gn;
                v.x = bp[0]; v.y = bp[ldb]; v.z = bp[2 * (long)ldb]; v.w = bp[3 * (long)ldb];
            }
            bR[j] = v;
        }
    };
    auto storeA = [&](_Float16* dst) {
        #pragma unroll
        for (int j = 0; j < ASLOT; ++j) {
            const int s = tid + j * 256;
            if (TRANSA) {
                const int kk = s >> 5;
                const int m  = (s & 31) * 4;
                const h4 h = to_h4(aR[j]);
                dst[(m + 0) * AROW + kk] = h.x;
                dst[(m + 1) * AROW + kk] = h.y;
                dst[(m + 2) * AROW + kk] = h.z;
                dst[(m + 3) * AROW + kk] = h.w;
            } else {
                const int m = s >> 4;
                const int k = (s & 15) * 4;
                *(h4*)(&dst[m * AROW + k]) = to_h4(aR[j]);
            }
        }
    };
    auto storeB = [&](_Float16* dst) {
        #pragma unroll
        for (int j = 0; j < BSLOT; ++j) {
            const int s = tid + j * 256;
            const int n = s & (BN - 1);
            const int k = (s >> 7) * 4;
            *(h4*)(&dst[n * BROW + k]) = to_h4(bR[j]);   // one ds_store_b64
        }
    };

    // ---- pipeline prologue ----
    loadA(0);
    loadB(0);
    storeA(sA[0]);
    storeB(sB[0]);
    int cur = 0;

    for (int k0 = 0; k0 < Ksz; k0 += BK) {
        __syncthreads();
        const bool more = (k0 + BK) < Ksz;
        if (more) {                 // issue next tile's global loads early
            loadA(k0 + BK);
            loadB(k0 + BK);
            if (k0 + 2 * BK < Ksz) {   // L2 prefetch two tiles ahead
                if (TRANSA)
                    __builtin_prefetch(A + (long)(k0 + 2 * BK + (tid >> 5)) * lda + blockM + (tid & 31) * 4, 0, 1);
                else
                    __builtin_prefetch(A + (long)(blockM + (tid >> 4)) * lda + k0 + 2 * BK + (tid & 15) * 4, 0, 1);
                __builtin_prefetch(Bm + (long)(k0 + 2 * BK + (tid >> 7) * 4) * ldb + blockN + (tid & 127), 0, 1);
            }
        }

        // ---- WMMA over the two 32-wide K windows of this tile ----
        #pragma unroll
        for (int kh = 0; kh < BK; kh += 32) {
            v16h af[2], bf[4];
            #pragma unroll
            for (int im = 0; im < 2; ++im) {
                const _Float16* base = &sA[cur][(wm * 32 + im * 16 + lmn) * AROW + kh];
                #pragma unroll
                for (int p = 0; p < 8; ++p) {
                    const int kb = (p & 3) * 2 + (p >> 2) * 16 + hi * 8;
                    af[im][2 * p]     = base[kb];
                    af[im][2 * p + 1] = base[kb + 1];
                }
            }
            #pragma unroll
            for (int in = 0; in < 4; ++in) {
                const _Float16* base = &sB[cur][(wn * 64 + in * 16 + lmn) * BROW + kh];
                #pragma unroll
                for (int p = 0; p < 8; ++p) {
                    const int kb = (p & 3) * 2 + (p >> 2) * 16 + hi * 8;
                    bf[in][2 * p]     = base[kb];
                    bf[in][2 * p + 1] = base[kb + 1];
                }
            }
            #pragma unroll
            for (int im = 0; im < 2; ++im)
                #pragma unroll
                for (int in = 0; in < 4; ++in)
                    acc[im][in] = __builtin_amdgcn_wmma_f32_16x16x32_f16(
                        false, af[im], false, bf[in],
                        (short)0, acc[im][in], false, false);
        }

        if (more) {                 // convert + stage into the other buffer
            storeA(sA[cur ^ 1]);
            storeB(sB[cur ^ 1]);
        }
        cur ^= 1;
    }

    // ---- epilogue: bias + optional elu(x)+1, store ----
    #pragma unroll
    for (int im = 0; im < 2; ++im) {
        #pragma unroll
        for (int in = 0; in < 4; ++in) {
            const int n = blockN + wn * 64 + in * 16 + lmn;
            if (n >= Nsz) continue;        // Nsz is 16-aligned -> lane-uniform per tile
            const float bv = bias ? bias[n] : 0.0f;
            #pragma unroll
            for (int r = 0; r < 8; ++r) {
                const int m = blockM + wm * 32 + im * 16 + r + 8 * hi;
                float v = acc[im][in][r] + bv;
                if (ACT == 1) v = (v > 0.0f) ? (v + 1.0f) : __expf(v);
                C[(long)m * ldc + n] = (TC)v;
            }
        }
    }
}

// k_sum[bh][m] = sum_l k_psi[bh][l][m]
__global__ __launch_bounds__(256)
void ksum_k(const _Float16* __restrict__ kpsi, float* __restrict__ ksum)
{
    const int bh = blockIdx.y;
    const int m  = threadIdx.x;            // Mk == 256 == blockDim.x
    const _Float16* p = kpsi + (long)bh * Lh * Mk + m;
    float s = 0.0f;
    for (int l = 0; l < Lh; ++l) s += (float)p[(long)l * Mk];
    ksum[bh * Mk + m] = s;
}

// out_pre[b][l][h*D + d] = qkv[bh][l][d] / (dot(q_phi[bh][l][:], ksum[bh][:]) + 1e-8)
__global__ __launch_bounds__(256)
void norm_k(const _Float16* __restrict__ qphi, const float* __restrict__ ksum,
            const float* __restrict__ qkv, float* __restrict__ out_pre)
{
    const int gwave = blockIdx.x * 8 + (threadIdx.x >> 5);   // one wave per (bh,l)
    const int lane  = threadIdx.x & 31;
    const int bh = gwave >> 12;            // / Lh
    const int l  = gwave & (Lh - 1);

    const _Float16* qp = qphi + ((long)bh * Lh + l) * Mk;
    const float*    ks = ksum + bh * Mk;

    float s = 0.0f;
    #pragma unroll
    for (int i = lane; i < Mk; i += 32) s += (float)qp[i] * ks[i];
    #pragma unroll
    for (int off = 16; off > 0; off >>= 1) s += __shfl_xor(s, off, 32);
    const float inv = 1.0f / (s + 1e-8f);

    const int b = bh / Hh, h = bh % Hh;
    const float* qv = qkv + ((long)bh * Lh + l) * Dh;
    float* op = out_pre + ((long)b * Lh + l) * Eh + h * Dh;
    #pragma unroll
    for (int d = lane; d < Dh; d += 32) op[d] = qv[d] * inv;
}

extern "C" void kernel_launch(void* const* d_in, const int* in_sizes, int n_in,
                              void* d_out, int out_size, void* d_ws, size_t ws_size,
                              hipStream_t stream)
{
    const float* x    = (const float*)d_in[0];
    const float* Wq   = (const float*)d_in[1];
    const float* bq   = (const float*)d_in[2];
    const float* Wk   = (const float*)d_in[3];
    const float* bk   = (const float*)d_in[4];
    const float* Wv   = (const float*)d_in[5];
    const float* bv   = (const float*)d_in[6];
    const float* Wphi = (const float*)d_in[7];
    const float* bphi = (const float*)d_in[8];
    const float* Wpsi = (const float*)d_in[9];
    const float* bpsi = (const float*)d_in[10];
    const float* Wo   = (const float*)d_in[11];
    const float* bo   = (const float*)d_in[12];
    float* out = (float*)d_out;

    const long BL = (long)Bb * Lh;                 // 16384
    const long BH = (long)Bb * Hh;                 // 64

    // workspace carve-up (bytes)
    char* ws = (char*)d_ws;
    size_t off = 0;
    auto take = [&](size_t bytes) { char* p = ws + off; off += (bytes + 255) & ~(size_t)255; return p; };
    float*    q       = (float*)   take(BL * Eh * sizeof(float));
    float*    kq      = (float*)   take(BL * Eh * sizeof(float));
    float*    vv      = (float*)   take(BL * Eh * sizeof(float));
    _Float16* qphi    = (_Float16*)take(BH * Lh * Mk * sizeof(_Float16));
    _Float16* kpsi    = (_Float16*)take(BH * Lh * Mk * sizeof(_Float16));
    float*    kv      = (float*)   take(BH * Mk * Dh * sizeof(float));
    float*    ksum    = (float*)   take(BH * Mk * sizeof(float));
    float*    qkv     = (float*)   take(BH * Lh * Dh * sizeof(float));
    float*    out_pre = (float*)   take(BL * Eh * sizeof(float));
    (void)ws_size;

    const dim3 blk(256);

    // 1-3) q,k,v = x @ W + b        [16384 x 1024 x 1024]
    {
        dim3 grd(Eh / BN, BL / BM, 1);
        gemm_wmma_k<float, float, false, 0><<<grd, blk, 0, stream>>>(
            x, 0, 0, Eh,  Wq, 0, 0, Eh,  bq,  q, 0, 0, Eh,  1, (int)BL, Eh, Eh);
        gemm_wmma_k<float, float, false, 0><<<grd, blk, 0, stream>>>(
            x, 0, 0, Eh,  Wk, 0, 0, Eh,  bk,  kq, 0, 0, Eh, 1, (int)BL, Eh, Eh);
        gemm_wmma_k<float, float, false, 0><<<grd, blk, 0, stream>>>(
            x, 0, 0, Eh,  Wv, 0, 0, Eh,  bv,  vv, 0, 0, Eh, 1, (int)BL, Eh, Eh);
    }

    // 4-5) q_phi = elu(q_h @ Wphi + bphi)+1 ; k_psi likewise   [per bh: 4096 x 256 x 64]
    {
        dim3 grd(Mk / BN, Lh / BM, (unsigned)BH);
        const long aO = (long)Lh * Eh, aI = Dh;                  // head view of q/k
        const long cO = (long)Hh * Lh * Mk, cI = (long)Lh * Mk;  // [bh][L][M]
        gemm_wmma_k<float, _Float16, false, 1><<<grd, blk, 0, stream>>>(
            q, aO, aI, Eh,  Wphi, 0, 0, Mk,  bphi,  qphi, cO, cI, Mk,  Hh, Lh, Mk, Dh);
        gemm_wmma_k<float, _Float16, false, 1><<<grd, blk, 0, stream>>>(
            kq, aO, aI, Eh, Wpsi, 0, 0, Mk,  bpsi,  kpsi, cO, cI, Mk,  Hh, Lh, Mk, Dh);
    }

    // 6) ksum[bh][m] = sum_l k_psi
    {
        dim3 grd(1, (unsigned)BH, 1);
        ksum_k<<<grd, blk, 0, stream>>>(kpsi, ksum);
    }

    // 7) kv[bh] = k_psi^T @ v_h      [per bh: 256 x 64 x 4096]
    {
        dim3 grd(1, Mk / BM, (unsigned)BH);
        const long aO = (long)Hh * Lh * Mk, aI = (long)Lh * Mk;  // k_psi batches (TRANSA)
        const long bO = (long)Lh * Eh, bI = Dh;                  // head view of v
        const long cO = (long)Hh * Mk * Dh, cI = (long)Mk * Dh;
        gemm_wmma_k<_Float16, float, true, 0><<<grd, blk, 0, stream>>>(
            kpsi, aO, aI, Mk,  vv, bO, bI, Eh,  nullptr,  kv, cO, cI, Dh,  Hh, Mk, Dh, Lh);
    }

    // 8) qkv[bh] = q_phi @ kv        [per bh: 4096 x 64 x 256]
    {
        dim3 grd(1, Lh / BM, (unsigned)BH);
        const long aO = (long)Hh * Lh * Mk, aI = (long)Lh * Mk;
        const long bO = (long)Hh * Mk * Dh, bI = (long)Mk * Dh;
        const long cO = (long)Hh * Lh * Dh, cI = (long)Lh * Dh;
        gemm_wmma_k<_Float16, float, false, 0><<<grd, blk, 0, stream>>>(
            qphi, aO, aI, Mk,  kv, bO, bI, Dh,  nullptr,  qkv, cO, cI, Dh,  Hh, Lh, Dh, Mk);
    }

    // 9) normalize + re-interleave heads -> out_pre [B,L,E]
    {
        const int nwaves = (int)(BH * Lh);          // 262144 rows
        dim3 grd(nwaves / 8, 1, 1);
        norm_k<<<grd, blk, 0, stream>>>(qphi, ksum, qkv, out_pre);
    }

    // 10) out = out_pre @ Wo + bo    [16384 x 1024 x 1024]
    {
        dim3 grd(Eh / BN, BL / BM, 1);
        gemm_wmma_k<float, float, false, 0><<<grd, blk, 0, stream>>>(
            out_pre, 0, 0, Eh,  Wo, 0, 0, Eh,  bo,  out, 0, 0, Eh,  1, (int)BL, Eh, Eh);
    }
}
